// MultiHeadAttention_21792664060369
// MI455X (gfx1250) — matrix operand order repaired and checked by reading, not compiled, not run on hardware
//
#include <hip/hip_runtime.h>

// ---------------------------------------------------------------------------
// MI455X (gfx1250) fused multi-head attention, bf16 WMMA pipeline + TDM
// (Tensor Data Mover) double-buffered LDS staging.
//   x[B,T,D] --(GEMM1)--> Q,K (scaled, [B,H,T,DH]) + V^T ([B,H,DH,T])  (bf16)
//   flash attention (online softmax, WMMA QK^T and P@V) --> ctx bf16 [B,T,D]
//   ctx --(GEMM2)--> out fp32 [B,T,D]
// ---------------------------------------------------------------------------

#define DEV __device__ __forceinline__

typedef __bf16 bf16;
typedef bf16  v16bf __attribute__((ext_vector_type(16)));
typedef bf16  v8bf  __attribute__((ext_vector_type(8)));
typedef bf16  v4bf  __attribute__((ext_vector_type(4)));
typedef float v8f   __attribute__((ext_vector_type(8)));

constexpr int B_  = 4;
constexpr int T_  = 2048;
constexpr int D_  = 1024;
constexpr int H_  = 16;
constexpr int DH_ = 64;
constexpr float FLOAT_MIN_ = -3.40282347e38f;

#if defined(__has_builtin)
#if __has_builtin(__builtin_amdgcn_tensor_load_to_lds) && \
    __has_builtin(__builtin_amdgcn_s_wait_tensorcnt)
#define USE_TDM 1
#endif
#endif
#ifndef USE_TDM
#define USE_TDM 0
#endif

DEV v8f wmma_bf16(v16bf a, v16bf b, v8f c) {
  // v_wmma_f32_16x16x32_bf16: D = A(16x32) * B(32x16) + C(16x16 f32)
  return __builtin_amdgcn_wmma_f32_16x16x32_bf16(false, a, false, b, (short)0, c,
                                                 false, false);
}

// A fragment (16x32 bf16, M x K), row-major source with leading dim lda.
DEV v16bf load_a_frag(const bf16* base, int lda, int row0, int k0, int lane) {
  int r = lane & 15, h = lane >> 4;
  const bf16* p = base + (size_t)(row0 + r) * lda + k0 + 16 * h;
  v8bf lo = *(const v8bf*)p;
  v8bf hi = *(const v8bf*)(p + 8);
  return __builtin_shufflevector(lo, hi, 0, 1, 2, 3, 4, 5, 6, 7,
                                 8, 9, 10, 11, 12, 13, 14, 15);
}

// B fragment (32x16 bf16, K x N) from an N-major ("transposed") source Bt[n][k].
DEV v16bf load_b_frag(const bf16* base, int ldb, int n0, int k0, int lane) {
  int r = lane & 15, h = lane >> 4;
  return *(const v16bf*)(base + (size_t)(n0 + r) * ldb + k0 + 16 * h);
}

#if USE_TDM
typedef unsigned int u32x4 __attribute__((ext_vector_type(4)));
typedef int          i32x4 __attribute__((ext_vector_type(4)));
typedef int          i32x8 __attribute__((ext_vector_type(8)));

// Generic "shared" pointers carry the LDS byte offset in the low 32 bits.
DEV unsigned int lds_addr_of(const void* p) {
  return (unsigned int)(uintptr_t)p;
}

// Issue one TDM 2D-tile load (bf16 elements) global -> LDS.
// tile_d0 elements per row (contiguous), tile_d1 rows, row pitch
// row_stride_elems; rows packed back-to-back in LDS at lds_byte.
DEV void tdm_load_tile_bf16(const bf16* gsrc, unsigned int lds_byte,
                            unsigned int tensor_d0, unsigned int tensor_d1,
                            unsigned int row_stride_elems,
                            unsigned int tile_d0, unsigned int tile_d1) {
  unsigned long long ga = (unsigned long long)(uintptr_t)gsrc;
  u32x4 g0;
  g0[0] = 1u;                                        // count=1, user mode
  g0[1] = lds_byte;                                  // lds_addr
  g0[2] = (unsigned int)ga;                          // global_addr[31:0]
  g0[3] = (unsigned int)((ga >> 32) & 0x1FFFFFFu) |  // global_addr[56:32]
          (2u << 30);                                // type=2 ("image")
  i32x8 g1;
  g1[0] = 1 << 16;                                   // data_size=1 (2 bytes)
  g1[1] = (int)((tensor_d0 & 0xFFFFu) << 16);        // tensor_dim0[15:0]
  g1[2] = (int)((tensor_d0 >> 16) | ((tensor_d1 & 0xFFFFu) << 16));
  g1[3] = (int)((tensor_d1 >> 16) | (tile_d0 << 16));// tile_dim0
  g1[4] = (int)tile_d1;                              // tile_dim1 (tile_dim2=0)
  g1[5] = (int)row_stride_elems;                     // tensor_dim0_stride[31:0]
  g1[6] = 0;                                         // stride hi / dim1_stride
  g1[7] = 0;
  i32x4 z4 = {0, 0, 0, 0};
#if __clang_major__ >= 23
  i32x8 z8 = {0, 0, 0, 0, 0, 0, 0, 0};
  __builtin_amdgcn_tensor_load_to_lds(g0, g1, z4, z4, z8, 0);
#else
  __builtin_amdgcn_tensor_load_to_lds(g0, g1, z4, z4, 0);
#endif
}
#endif  // USE_TDM

// ---------------------------------------------------------------------------
// Conversion helpers
// ---------------------------------------------------------------------------
__global__ __launch_bounds__(256) void convert_bf16_kernel(
    const float* __restrict__ in, bf16* __restrict__ out, int n4) {
  int i = blockIdx.x * 256 + threadIdx.x;
  if (i < n4) {
    float4 v = ((const float4*)in)[i];
    v4bf o;
    o[0] = (bf16)v.x; o[1] = (bf16)v.y; o[2] = (bf16)v.z; o[3] = (bf16)v.w;
    ((v4bf*)out)[i] = o;
  }
}

// out[n][k] = (bf16) in[k][n]   (K x N fp32 -> N x K bf16)
__global__ __launch_bounds__(256) void transpose_convert_kernel(
    const float* __restrict__ in, bf16* __restrict__ out, int K, int N) {
  __shared__ float tile[32][33];
  int kx = blockIdx.x * 32, nx = blockIdx.y * 32;
  int tx = threadIdx.x, ty = threadIdx.y;
  #pragma unroll
  for (int i = ty; i < 32; i += 8)
    tile[i][tx] = in[(size_t)(kx + i) * N + nx + tx];
  __syncthreads();
  #pragma unroll
  for (int i = ty; i < 32; i += 8)
    out[(size_t)(nx + i) * K + kx + tx] = (bf16)tile[tx][i];
}

// ---------------------------------------------------------------------------
// Shared GEMM mainloop: C(64x64) per block, 4 waves, K-step 32, double-
// buffered LDS tiles fed by TDM (fallback: cooperative vector loads).
// Result: per-wave f[4] C fragments.
// ---------------------------------------------------------------------------
struct GemmFrag { v8f f[4]; };

template <int K>
DEV GemmFrag gemm_mainloop(const bf16* __restrict__ A,
                           const bf16* __restrict__ Bt, int m0, int n0,
                           bf16 (*a_lds)[64 * 32], bf16 (*b_lds)[64 * 32],
                           int tid) {
  constexpr int TM = 64, TN = 64, TK = 32;
  int wave = tid >> 5, lane = tid & 31;

  GemmFrag g;
  #pragma unroll
  for (int i = 0; i < 4; i++)
    #pragma unroll
    for (int j = 0; j < 8; j++) g.f[i][j] = 0.f;

  constexpr int NK = K / TK;
#if USE_TDM
  if (wave == 0) {
    tdm_load_tile_bf16(&A[(size_t)m0 * K], lds_addr_of(a_lds[0]), K, TM, K, TK, TM);
    tdm_load_tile_bf16(&Bt[(size_t)n0 * K], lds_addr_of(b_lds[0]), K, TN, K, TK, TN);
  }
#endif
  for (int it = 0; it < NK; ++it) {
    int buf = it & 1;
    __syncthreads();  // all waves done with buf^1 (prefetch target)
#if USE_TDM
    if (wave == 0) {
      if (it + 1 < NK) {
        int kb = (it + 1) * TK;
        tdm_load_tile_bf16(&A[(size_t)m0 * K + kb], lds_addr_of(a_lds[buf ^ 1]),
                           K, TM, K, TK, TM);
        tdm_load_tile_bf16(&Bt[(size_t)n0 * K + kb], lds_addr_of(b_lds[buf ^ 1]),
                           K, TN, K, TK, TN);
        __builtin_amdgcn_s_wait_tensorcnt(2);  // current buf's 2 loads done
      } else {
        __builtin_amdgcn_s_wait_tensorcnt(0);
      }
    }
#else
    {
      int kb = it * TK;
      #pragma unroll
      for (int c = tid; c < (TM * TK) / 8; c += 128) {
        int row = c >> 2, col = (c & 3) * 8;
        *(v8bf*)&a_lds[buf][row * TK + col] =
            *(const v8bf*)&A[(size_t)(m0 + row) * K + kb + col];
        *(v8bf*)&b_lds[buf][row * TK + col] =
            *(const v8bf*)&Bt[(size_t)(n0 + row) * K + kb + col];
      }
    }
#endif
    __syncthreads();  // buf resident for all waves
    v16bf af = load_a_frag(a_lds[buf], TK, wave * 16, 0, lane);
    #pragma unroll
    for (int nt = 0; nt < 4; nt++)
      g.f[nt] = wmma_bf16(af, load_b_frag(b_lds[buf], TK, nt * 16, 0, lane),
                          g.f[nt]);
  }
  return g;
}

// ---------------------------------------------------------------------------
// GEMM1: qkv = x @ W_qkv + b ; scatter Q(scaled)/K -> [B,H,T,DH],
// V -> [B,H,DH,T] (bf16).
// ---------------------------------------------------------------------------
__global__ __launch_bounds__(128) void gemm_qkv_kernel(
    const bf16* __restrict__ A, const bf16* __restrict__ Bt,
    const float* __restrict__ bias, bf16* __restrict__ Qb,
    bf16* __restrict__ Kb, bf16* __restrict__ Vtb) {
  __shared__ __align__(32) bf16 a_lds[2][64 * 32];
  __shared__ __align__(32) bf16 b_lds[2][64 * 32];
  int m0 = blockIdx.x * 64, n0 = blockIdx.y * 64;
  int tid = threadIdx.x, wave = tid >> 5, lane = tid & 31;

  GemmFrag g = gemm_mainloop<D_>(A, Bt, m0, n0, a_lds, b_lds, tid);

  int r = lane & 15, h = lane >> 4;
  #pragma unroll
  for (int nt = 0; nt < 4; nt++) {
    int n = n0 + nt * 16 + r;
    float bv = bias[n];
    int sec = n >> 10;  // 0:q 1:k 2:v
    int d = n & 1023;
    int head = d >> 6, dh = d & 63;
    #pragma unroll
    for (int j = 0; j < 8; j++) {
      int row = m0 + wave * 16 + 8 * h + j;
      int b = row >> 11, t = row & (T_ - 1);
      float val = g.f[nt][j] + bv;
      size_t hidx = ((size_t)(b * H_ + head) * T_ + t) * DH_ + dh;
      if (sec == 0)
        Qb[hidx] = (bf16)(val * 0.125f);  // DH^-0.5
      else if (sec == 1)
        Kb[hidx] = (bf16)val;
      else
        Vtb[((size_t)(b * H_ + head) * DH_ + dh) * (size_t)T_ + t] = (bf16)val;
    }
  }
}

// ---------------------------------------------------------------------------
// Flash attention: block = (b,h, 64 q rows); 4 waves, each a 16-row q tile.
// K/V tiles double-buffered in LDS via TDM.
// ---------------------------------------------------------------------------
__global__ __launch_bounds__(128) void flash_attn_kernel(
    const bf16* __restrict__ Qb, const bf16* __restrict__ Kb,
    const bf16* __restrict__ Vtb, const float* __restrict__ mask,
    bf16* __restrict__ ctxb) {
  constexpr int BQ = 64, BK = 64, PLD = 72;
  __shared__ __align__(32) bf16 klds[2][BK * DH_];   // [tk_local][dh]
  __shared__ __align__(32) bf16 vlds[2][DH_ * BK];   // [dh][tk_local]
  __shared__ __align__(32) bf16 plds[4][16 * PLD];   // per-wave P tile

  int bh = blockIdx.y;
  int b = bh >> 4, head = bh & 15;
  int q0 = blockIdx.x * BQ;
  int tid = threadIdx.x, wave = tid >> 5, lane = tid & 31;
  int r = lane & 15, h = lane >> 4;

  const bf16* Qh = Qb + (size_t)bh * T_ * DH_;
  const bf16* Kh = Kb + (size_t)bh * T_ * DH_;
  const bf16* Vh = Vtb + (size_t)bh * DH_ * T_;
  const float* mb = mask + (size_t)b * T_ * T_;

  v16bf qf0 = load_a_frag(Qh, DH_, q0 + wave * 16, 0, lane);
  v16bf qf1 = load_a_frag(Qh, DH_, q0 + wave * 16, 32, lane);

  float m_run[8], l_run[8];
  v8f o[4];
  #pragma unroll
  for (int j = 0; j < 8; j++) { m_run[j] = FLOAT_MIN_; l_run[j] = 0.f; }
  #pragma unroll
  for (int ot = 0; ot < 4; ot++)
    #pragma unroll
    for (int j = 0; j < 8; j++) o[ot][j] = 0.f;

  bf16* pw = &plds[wave][0];

  constexpr int NIT = T_ / BK;
#if USE_TDM
  if (wave == 0) {
    tdm_load_tile_bf16(Kh, lds_addr_of(klds[0]), DH_, T_, DH_, DH_, BK);
    tdm_load_tile_bf16(Vh, lds_addr_of(vlds[0]), T_, DH_, T_, BK, DH_);
  }
#endif
  for (int it = 0; it < NIT; ++it) {
    int tk = it * BK;
    int buf = it & 1;
    __syncthreads();  // everyone done with buf^1 (prefetch target)
#if USE_TDM
    if (wave == 0) {
      if (it + 1 < NIT) {
        int tkn = (it + 1) * BK;
        tdm_load_tile_bf16(Kh + (size_t)tkn * DH_, lds_addr_of(klds[buf ^ 1]),
                           DH_, T_, DH_, DH_, BK);
        tdm_load_tile_bf16(Vh + tkn, lds_addr_of(vlds[buf ^ 1]),
                           T_, DH_, T_, BK, DH_);
        __builtin_amdgcn_s_wait_tensorcnt(2);
      } else {
        __builtin_amdgcn_s_wait_tensorcnt(0);
      }
    }
#else
    #pragma unroll
    for (int c = tid; c < (BK * DH_) / 8; c += 128) {
      int row = c >> 3, col = (c & 7) * 8;
      *(v8bf*)&klds[buf][row * DH_ + col] =
          *(const v8bf*)&Kh[(size_t)(tk + row) * DH_ + col];
      *(v8bf*)&vlds[buf][row * BK + col] =
          *(const v8bf*)&Vh[(size_t)row * T_ + tk + col];
    }
#endif
    __syncthreads();  // buf resident

    // S = Q K^T for four 16x16 n-subtiles
    v8f s[4];
    #pragma unroll
    for (int nt = 0; nt < 4; nt++) {
      v8f acc;
      #pragma unroll
      for (int j = 0; j < 8; j++) acc[j] = 0.f;
      acc = wmma_bf16(qf0, load_b_frag(klds[buf], DH_, nt * 16, 0, lane), acc);
      acc = wmma_bf16(qf1, load_b_frag(klds[buf], DH_, nt * 16, 32, lane), acc);
      s[nt] = acc;
    }

    // mask + row-wise max (row m = 8h + j lives on lanes r=0..15 of half h)
    float tmax[8];
    #pragma unroll
    for (int j = 0; j < 8; j++) tmax[j] = FLOAT_MIN_;
    #pragma unroll
    for (int nt = 0; nt < 4; nt++) {
      int kcol = tk + nt * 16 + r;
      #pragma unroll
      for (int j = 0; j < 8; j++) {
        int qrow = q0 + wave * 16 + 8 * h + j;
        float m = mb[(size_t)qrow * T_ + kcol];
        float val = s[nt][j] * m + (1.f - m) * FLOAT_MIN_;
        s[nt][j] = val;
        tmax[j] = fmaxf(tmax[j], val);
      }
    }
    #pragma unroll
    for (int off = 8; off >= 1; off >>= 1)
      #pragma unroll
      for (int j = 0; j < 8; j++)
        tmax[j] = fmaxf(tmax[j], __shfl_xor(tmax[j], off, 32));

    // online softmax update
    float corr[8], psum[8];
    #pragma unroll
    for (int j = 0; j < 8; j++) {
      float mnew = fmaxf(m_run[j], tmax[j]);
      corr[j] = __expf(m_run[j] - mnew);
      m_run[j] = mnew;
      psum[j] = 0.f;
    }
    #pragma unroll
    for (int nt = 0; nt < 4; nt++)
      #pragma unroll
      for (int j = 0; j < 8; j++) {
        float p = __expf(s[nt][j] - m_run[j]);
        s[nt][j] = p;
        psum[j] += p;
      }
    #pragma unroll
    for (int off = 8; off >= 1; off >>= 1)
      #pragma unroll
      for (int j = 0; j < 8; j++)
        psum[j] += __shfl_xor(psum[j], off, 32);
    #pragma unroll
    for (int j = 0; j < 8; j++) l_run[j] = l_run[j] * corr[j] + psum[j];
    #pragma unroll
    for (int ot = 0; ot < 4; ot++)
      #pragma unroll
      for (int j = 0; j < 8; j++) o[ot][j] *= corr[j];

    // re-shape P (C-fragment) -> A-fragment via per-wave LDS tile
    #pragma unroll
    for (int nt = 0; nt < 4; nt++)
      #pragma unroll
      for (int j = 0; j < 8; j++)
        pw[(8 * h + j) * PLD + nt * 16 + r] = (bf16)s[nt][j];
    __syncthreads();

    // O += P @ V
    #pragma unroll
    for (int ot = 0; ot < 4; ot++) {
      o[ot] = wmma_bf16(load_a_frag(pw, PLD, 0, 0, lane),
                        load_b_frag(vlds[buf], BK, ot * 16, 0, lane), o[ot]);
      o[ot] = wmma_bf16(load_a_frag(pw, PLD, 0, 32, lane),
                        load_b_frag(vlds[buf], BK, ot * 16, 32, lane), o[ot]);
    }
  }

  // normalize and write ctx[b][t][head*64 + dh] as bf16
  #pragma unroll
  for (int ot = 0; ot < 4; ot++)
    #pragma unroll
    for (int j = 0; j < 8; j++) {
      int t = q0 + wave * 16 + 8 * h + j;
      float val = o[ot][j] / l_run[j];
      ctxb[((size_t)b * T_ + t) * D_ + head * DH_ + ot * 16 + r] = (bf16)val;
    }
}

// ---------------------------------------------------------------------------
// GEMM2: out = ctx @ W_out + b_out  (fp32 output)
// ---------------------------------------------------------------------------
__global__ __launch_bounds__(128) void gemm_out_kernel(
    const bf16* __restrict__ A, const bf16* __restrict__ Bt,
    const float* __restrict__ bias, float* __restrict__ Cout) {
  __shared__ __align__(32) bf16 a_lds[2][64 * 32];
  __shared__ __align__(32) bf16 b_lds[2][64 * 32];
  int m0 = blockIdx.x * 64, n0 = blockIdx.y * 64;
  int tid = threadIdx.x, wave = tid >> 5, lane = tid & 31;

  GemmFrag g = gemm_mainloop<D_>(A, Bt, m0, n0, a_lds, b_lds, tid);

  int r = lane & 15, h = lane >> 4;
  #pragma unroll
  for (int nt = 0; nt < 4; nt++) {
    int n = n0 + nt * 16 + r;
    float bv = bias[n];
    #pragma unroll
    for (int j = 0; j < 8; j++) {
      int row = m0 + wave * 16 + 8 * h + j;
      Cout[(size_t)row * D_ + n] = g.f[nt][j] + bv;
    }
  }
}

// ---------------------------------------------------------------------------
extern "C" void kernel_launch(void* const* d_in, const int* in_sizes, int n_in,
                              void* d_out, int out_size, void* d_ws,
                              size_t ws_size, hipStream_t stream) {
  (void)in_sizes; (void)n_in; (void)out_size; (void)ws_size;
  const float* x     = (const float*)d_in[0];
  const float* W_qkv = (const float*)d_in[1];
  const float* b_qkv = (const float*)d_in[2];
  const float* W_out = (const float*)d_in[3];
  const float* b_out = (const float*)d_in[4];
  const float* mask  = (const float*)d_in[5];
  float* out = (float*)d_out;

  char* w = (char*)d_ws;
  size_t off = 0;
  auto carve = [&](size_t bytes) -> void* {
    off = (off + 255) & ~(size_t)255;
    void* p = w + off;
    off += bytes;
    return p;
  };
  bf16* xb    = (bf16*)carve((size_t)B_ * T_ * D_ * 2);
  bf16* WqkvT = (bf16*)carve((size_t)3 * D_ * D_ * 2);
  bf16* WoutT = (bf16*)carve((size_t)D_ * D_ * 2);
  bf16* Qb    = (bf16*)carve((size_t)B_ * H_ * T_ * DH_ * 2);
  bf16* Kb    = (bf16*)carve((size_t)B_ * H_ * T_ * DH_ * 2);
  bf16* Vtb   = (bf16*)carve((size_t)B_ * H_ * DH_ * T_ * 2);
  bf16* ctxb  = (bf16*)carve((size_t)B_ * T_ * D_ * 2);

  int nx4 = (B_ * T_ * D_) / 4;
  convert_bf16_kernel<<<nx4 / 256, 256, 0, stream>>>(x, xb, nx4);
  transpose_convert_kernel<<<dim3(D_ / 32, 3 * D_ / 32), dim3(32, 8), 0,
                             stream>>>(W_qkv, WqkvT, D_, 3 * D_);
  transpose_convert_kernel<<<dim3(D_ / 32, D_ / 32), dim3(32, 8), 0, stream>>>(
      W_out, WoutT, D_, D_);
  gemm_qkv_kernel<<<dim3(B_ * T_ / 64, 3 * D_ / 64), 128, 0, stream>>>(
      xb, WqkvT, b_qkv, Qb, Kb, Vtb);
  flash_attn_kernel<<<dim3(T_ / 64, B_ * H_), 128, 0, stream>>>(Qb, Kb, Vtb,
                                                                mask, ctxb);
  gemm_out_kernel<<<dim3(B_ * T_ / 64, D_ / 64), 128, 0, stream>>>(
      ctxb, WoutT, b_out, out);
}
